// Block_60232621359414
// MI455X (gfx1250) — compile-verified, matrix operand order checked by reading, not run on hardware
//
#include <hip/hip_runtime.h>
#include <hip/hip_bf16.h>

// ---------------------------------------------------------------------------
// Transformer block (attention + top-2 MoE) for MI455X / gfx1250, wave32.
// All GEMMs + attention on v_wmma_f32_16x16x32_bf16, fp32 accumulate.
// GEMM A-tiles and attention V-tiles staged into LDS by the Tensor Data
// Mover (tensor_load_to_lds + s_wait_tensorcnt) when the builtin exists;
// falls back to global_load_b128 + ds_store staging otherwise.
// ---------------------------------------------------------------------------

#define B_   4
#define T_   1024
#define C_   1024
#define H_   16
#define D_   64
#define E_   8
#define HID_ 4096
#define N_   (B_ * T_)     // 4096 tokens

#ifndef __has_builtin
#  define __has_builtin(x) 0
#endif
#if defined(__HIP_DEVICE_COMPILE__) && defined(__gfx1250__) && \
    __has_builtin(__builtin_amdgcn_tensor_load_to_lds)
#  define ATHENA_TDM 1
#  if __has_include(<hip/amd_detail/amd_gfx1250_TDM.h>)
#    define ATHENA_TDM_6ARG 1     // therock-10.0 headers -> 6-arg builtin
#  else
#    define ATHENA_TDM_6ARG 0     // ROCm 7.2 clang-22 -> 5-arg builtin
#  endif
#else
#  define ATHENA_TDM 0
#endif

typedef __attribute__((ext_vector_type(16))) __bf16 v16bf;
typedef __attribute__((ext_vector_type(8)))  float  v8f;
typedef __attribute__((ext_vector_type(4)))  unsigned int v4u;
typedef __attribute__((ext_vector_type(8)))  int v8i;
typedef __attribute__((ext_vector_type(4)))  int v4i;

union FragBF { v16bf v; uint4 q[2]; __bf16 h[16]; };

__device__ __forceinline__ __bf16 f2bf(float f) {
  union { float f; unsigned u; } a; a.f = f;
  unsigned r = a.u + 0x7FFFu + ((a.u >> 16) & 1u);
  union { unsigned short s; __bf16 b; } o; o.s = (unsigned short)(r >> 16);
  return o.b;
}
__device__ __forceinline__ float bf2f(__bf16 b) {
  union { unsigned short s; __bf16 b; } i; i.b = b;
  union { unsigned u; float f; } o; o.u = ((unsigned)i.s) << 16;
  return o.f;
}

#if ATHENA_TDM
// ---------------------------------------------------------------------------
// Issue one 2D TDM tile load: global (strided rows) -> LDS (tight rows).
// D# per CDNA5 ISA §8.3-8.5. data_size=2B, no padding, no iteration,
// workgroup_mask=0 (not in a cluster). Issued by one wave; EXEC ignored.
// ---------------------------------------------------------------------------
__device__ __forceinline__ void tdm_load_2d(const void* gptr, unsigned lds_off,
                                            int tile_w, int tile_h,
                                            int stride_elems,
                                            int tensor_w, int tensor_h) {
  unsigned long long ga = (unsigned long long)(size_t)gptr;
  // group0: [1:0]=count=1, [63:32]=lds_addr, [120:64]=global_addr, [127:126]=type 2
  v4u g0 = { 1u,
             lds_off,
             (unsigned)(ga & 0xffffffffu),
             ((unsigned)((ga >> 32) & 0x01ffffffu)) | (2u << 30) };
  // group1 (bit offsets within 256b):
  //  [17:16]=data_size=1 (2B)   [79:48]=tensor_dim0   [111:80]=tensor_dim1
  //  [127:112]=tile_dim0  [143:128]=tile_dim1  [159:144]=tile_dim2=0
  //  [207:160]=tensor_dim0_stride  [255:208]=tensor_dim1_stride=0
  unsigned tw = (unsigned)tensor_w, th = (unsigned)tensor_h;
  v8i g1 = { (int)(1u << 16),
             (int)((tw & 0xffffu) << 16),
             (int)(((tw >> 16) & 0xffffu) | ((th & 0xffffu) << 16)),
             (int)(((th >> 16) & 0xffffu) | (((unsigned)tile_w & 0xffffu) << 16)),
             (int)((unsigned)tile_h & 0xffffu),
             (int)(unsigned)stride_elems,
             0, 0 };
  v4i gz = { 0, 0, 0, 0 };
#if ATHENA_TDM_6ARG
  v8i gz8 = { 0, 0, 0, 0, 0, 0, 0, 0 };
  __builtin_amdgcn_tensor_load_to_lds(g0, g1, gz, gz, gz8, 0);
#else
  __builtin_amdgcn_tensor_load_to_lds(g0, g1, gz, gz, 0);
#endif
}
#endif  // ATHENA_TDM

// ---------------------------------------------------------------------------
// Tiled bf16 WMMA GEMM: C[M,N] = A[M,K] @ B[K,N] (+bias, + mode epilogue)
// Block tile 64x64, K-step 32. 4 waves, each wave 32x32 via 2x2 WMMA frags.
// MODE 0: out_bf = acc + bias                     (QKV)
// MODE 1: out_f  = residual + acc + bias          (proj + residual)
// MODE 2: out_bf = gelu(acc + bias)               (FFN up)
// MODE 3: out_f += scale_col[m] * (acc + bias)    (FFN down, MoE combine)
// ---------------------------------------------------------------------------
#define BM 64
#define BN 64
#define BK 32
#define LDA_T 32  // A tile LDS row stride in halves (tight: matches TDM layout)
#define LDT 40    // B tile LDS row stride in halves (80B, 16B-aligned)

template<int MODE>
__global__ __launch_bounds__(128)
void gemm_bf16_kernel(const __bf16* __restrict__ A, int lda,
                      const __bf16* __restrict__ Bm, int ldb,
                      const float* __restrict__ bias,
                      const float* __restrict__ residual,
                      const float* __restrict__ scale_col, int scale_stride,
                      __bf16* __restrict__ out_bf, float* __restrict__ out_f,
                      int ldo, int M, int N, int K) {
  __shared__ __bf16 sA[BM * LDA_T];
  __shared__ __bf16 sB[BN * LDT];   // B tile stored transposed: [n][k]
  const int tid  = threadIdx.x;
  const int lane = tid & 31, wave = tid >> 5;
  const int bm = blockIdx.y * BM, bn = blockIdx.x * BN;
  const int wm = (wave >> 1) * 32, wn = (wave & 1) * 32;
  const int row = lane & 15, kb = (lane >> 4) * 8;

  v8f acc[2][2] = {};

  for (int k0 = 0; k0 < K; k0 += BK) {
    __syncthreads();
#if ATHENA_TDM
    // Tensor Data Mover: async DMA of the strided 64x32 A tile into LDS.
    if (wave == 0) {
      tdm_load_2d(A + (size_t)bm * lda + k0, (unsigned)(size_t)&sA[0],
                  BK, BM, lda, K, M);
      __builtin_amdgcn_s_wait_tensorcnt((short)0);
    }
#else
    // Fallback: 16B vector loads -> LDS
#pragma unroll
    for (int i = 0; i < 2; ++i) {
      int c = tid + 128 * i;
      int r = c >> 2, c8 = (c & 3) * 8;
      *(uint4*)&sA[r * LDA_T + c8] =
          *(const uint4*)(A + (size_t)(bm + r) * lda + k0 + c8);
    }
#endif
    // Stage B tile transposed (LDS[n][k])
#pragma unroll
    for (int i = 0; i < 2; ++i) {
      int c = tid + 128 * i;
      int kk = c >> 3, n8 = (c & 7) * 8;
      uint4 d = *(const uint4*)(Bm + (size_t)(k0 + kk) * ldb + bn + n8);
      const __bf16* hp = (const __bf16*)&d;
#pragma unroll
      for (int j = 0; j < 8; ++j) sB[(n8 + j) * LDT + kk] = hp[j];
    }
    // Prefetch next K-tile (global_prefetch_b8)
    if (k0 + BK < K) {
      __builtin_prefetch(A + (size_t)(bm + (tid >> 2)) * lda + (k0 + BK) + (tid & 3) * 8, 0, 3);
      __builtin_prefetch(Bm + (size_t)(k0 + BK + (tid >> 3)) * ldb + bn + (tid & 7) * 8, 0, 3);
    }
    __syncthreads();

    // Fragments per the 16-bit A-matrix layout: lane = M (or N), kbase=(lane>>4)*8
    FragBF fa[2], fb[2];
#pragma unroll
    for (int i = 0; i < 2; ++i) {
      const __bf16* pa = &sA[(wm + i * 16 + row) * LDA_T + kb];
      fa[i].q[0] = *(const uint4*)pa;
      fa[i].q[1] = *(const uint4*)(pa + 16);
      const __bf16* pb = &sB[(wn + i * 16 + row) * LDT + kb];
      fb[i].q[0] = *(const uint4*)pb;
      fb[i].q[1] = *(const uint4*)(pb + 16);
    }
#pragma unroll
    for (int i = 0; i < 2; ++i)
#pragma unroll
      for (int j = 0; j < 2; ++j)
        acc[i][j] = __builtin_amdgcn_wmma_f32_16x16x32_bf16(
            false, fa[i].v, false, fb[j].v, (short)0, acc[i][j], false, false);
  }

  // Epilogue. C-layout: lanes 0-15 -> N=lane, M=v; lanes 16-31 -> M=8+v
  const int nl = lane & 15, mh = (lane >> 4) * 8;
#pragma unroll
  for (int i = 0; i < 2; ++i)
#pragma unroll
    for (int j = 0; j < 2; ++j)
#pragma unroll
      for (int v = 0; v < 8; ++v) {
        int gm = bm + wm + i * 16 + mh + v;
        int gn = bn + wn + j * 16 + nl;
        float val = acc[i][j][v] + (bias ? bias[gn] : 0.f);
        size_t o = (size_t)gm * ldo + gn;
        if (MODE == 0) {
          out_bf[o] = f2bf(val);
        } else if (MODE == 1) {
          out_f[o] = residual[o] + val;
        } else if (MODE == 2) {
          float g = 0.5f * val * (1.f + erff(val * 0.70710678118654752f));
          out_bf[o] = f2bf(g);
        } else {
          float sc = scale_col[(size_t)gm * scale_stride];
          out_f[o] += sc * val;
        }
      }
}

// ---------------------------------------------------------------------------
// Flash attention, causal. Block = (qtile of 64 rows, head, batch), 4 waves.
// Each wave: 16 q rows. KV streamed in tiles of 32.
// S = Q@K^T via WMMA (contraction over D=64 in 2 steps), online softmax,
// P routed through per-wave LDS to A-layout, O += P@V via WMMA.
// V tile staged by TDM when available.
// ---------------------------------------------------------------------------
#if ATHENA_TDM
#define VLD 64   // TDM writes the tile tight: 64 halves per row
#else
#define VLD 72   // padded for the fallback path
#endif
#define PLD 40

__global__ __launch_bounds__(128)
void attn_kernel(const __bf16* __restrict__ qkv, __bf16* __restrict__ y) {
  __shared__ __bf16 sV[32 * VLD];
  __shared__ __bf16 sP[4 * 16 * PLD];
  const int tid = threadIdx.x;
  const int lane = tid & 31, wave = tid >> 5;
  const int qtile = blockIdx.x, h = blockIdx.y, b = blockIdx.z;
  const size_t bT = (size_t)b * T_;
  const int qb = qtile * 64 + wave * 16;
  const int nl = lane & 15, kb = (lane >> 4) * 8, mh = (lane >> 4) * 8;

  // Q fragments (A-layout), loaded straight from global (contiguous 16B runs)
  FragBF qf[2];
  {
    const __bf16* qp = qkv + (bT + qb + nl) * (size_t)(3 * C_) + h * D_;
#pragma unroll
    for (int f = 0; f < 2; ++f) {
      qf[f].q[0] = *(const uint4*)(qp + f * 32 + kb);
      qf[f].q[1] = *(const uint4*)(qp + f * 32 + 16 + kb);
    }
  }

  v8f o[4] = {};
  float mrow[8], lrow[8];
#pragma unroll
  for (int v = 0; v < 8; ++v) { mrow[v] = -1e30f; lrow[v] = 0.f; }
  const float SC = 0.125f;   // 1/sqrt(D)
  __bf16* sPw = &sP[wave * 16 * PLD];

  const int kv_lim = qtile * 64 + 64;
  for (int kv0 = 0; kv0 < kv_lim; kv0 += 32) {
    __syncthreads();
#if ATHENA_TDM
    if (wave == 0) {
      tdm_load_2d(qkv + (bT + kv0) * (size_t)(3 * C_) + 2 * C_ + h * D_,
                  (unsigned)(size_t)&sV[0], D_, 32, 3 * C_, 3 * C_, T_);
      __builtin_amdgcn_s_wait_tensorcnt((short)0);
    }
#else
    // Stage V tile [32 kv][64 d]
#pragma unroll
    for (int i = 0; i < 2; ++i) {
      int c = tid + 128 * i;
      int kk = c >> 3, d8 = (c & 7) * 8;
      *(uint4*)&sV[kk * VLD + d8] =
          *(const uint4*)(qkv + (bT + kv0 + kk) * (size_t)(3 * C_) + 2 * C_ + h * D_ + d8);
    }
#endif
    __syncthreads();

    // S = Q @ K^T : 2 N-frags (kv groups of 16) x 2 K-steps over D
    v8f sacc[2] = {};
#pragma unroll
    for (int g = 0; g < 2; ++g) {
      const __bf16* kp = qkv + (bT + kv0 + g * 16 + nl) * (size_t)(3 * C_) + C_ + h * D_;
#pragma unroll
      for (int f = 0; f < 2; ++f) {
        FragBF kf;
        kf.q[0] = *(const uint4*)(kp + f * 32 + kb);
        kf.q[1] = *(const uint4*)(kp + f * 32 + 16 + kb);
        sacc[g] = __builtin_amdgcn_wmma_f32_16x16x32_bf16(
            false, qf[f].v, false, kf.v, (short)0, sacc[g], false, false);
      }
    }

    // Online softmax update (per-row stats live redundantly across 16 lanes)
#pragma unroll
    for (int v = 0; v < 8; ++v) {
      int tq = qb + mh + v;
      float s0 = sacc[0][v] * SC; if (kv0 + nl > tq)      s0 = -1e30f;
      float s1 = sacc[1][v] * SC; if (kv0 + 16 + nl > tq) s1 = -1e30f;
      float rm = fmaxf(s0, s1);
#pragma unroll
      for (int m = 8; m >= 1; m >>= 1) rm = fmaxf(rm, __shfl_xor(rm, m, 32));
      float nm = fmaxf(mrow[v], rm);
      float p0 = __expf(s0 - nm), p1 = __expf(s1 - nm);
      float rs = p0 + p1;
#pragma unroll
      for (int m = 8; m >= 1; m >>= 1) rs += __shfl_xor(rs, m, 32);
      float alpha = __expf(mrow[v] - nm);
      lrow[v] = lrow[v] * alpha + rs;
      mrow[v] = nm;
#pragma unroll
      for (int j = 0; j < 4; ++j) o[j][v] *= alpha;
      sPw[(mh + v) * PLD + nl]      = f2bf(p0);
      sPw[(mh + v) * PLD + 16 + nl] = f2bf(p1);
    }
    // Per-wave LDS round-trip: ensure DS stores landed before fragment reload
    asm volatile("s_wait_dscnt 0" ::: "memory");

    // P fragment (A-layout) from per-wave LDS region
    FragBF pf;
    {
      const __bf16* pp = &sPw[nl * PLD + kb];
      pf.q[0] = *(const uint4*)pp;
      pf.q[1] = *(const uint4*)(pp + 16);
    }
    // O += P @ V  : 4 N-frags over D
#pragma unroll
    for (int j = 0; j < 4; ++j) {
      FragBF vf;
#pragma unroll
      for (int u = 0; u < 8; ++u) {
        vf.h[u]     = sV[(kb + u) * VLD + j * 16 + nl];
        vf.h[8 + u] = sV[(16 + kb + u) * VLD + j * 16 + nl];
      }
      o[j] = __builtin_amdgcn_wmma_f32_16x16x32_bf16(
          false, pf.v, false, vf.v, (short)0, o[j], false, false);
    }
  }

  // Normalize + store y (bf16, feeds proj GEMM)
#pragma unroll
  for (int v = 0; v < 8; ++v) {
    float inv = 1.f / lrow[v];
    int tq = qb + mh + v;
    __bf16* yp = y + (bT + tq) * (size_t)C_ + h * D_;
#pragma unroll
    for (int j = 0; j < 4; ++j) yp[j * 16 + nl] = f2bf(o[j][v] * inv);
  }
}

// ---------------------------------------------------------------------------
// LayerNorm + pack to bf16 (one block per token row)
// ---------------------------------------------------------------------------
__global__ __launch_bounds__(256)
void ln_pack_kernel(const float* __restrict__ x, const float* __restrict__ g,
                    const float* __restrict__ bb, __bf16* __restrict__ out) {
  __shared__ float red[256];
  const int row = blockIdx.x;
  const float* xr = x + (size_t)row * C_;
  float s = 0.f, s2 = 0.f;
  for (int c = threadIdx.x; c < C_; c += 256) { float v = xr[c]; s += v; s2 += v * v; }
  red[threadIdx.x] = s; __syncthreads();
  for (int off = 128; off > 0; off >>= 1) {
    if (threadIdx.x < off) red[threadIdx.x] += red[threadIdx.x + off];
    __syncthreads();
  }
  float mean = red[0] * (1.f / C_);
  __syncthreads();
  red[threadIdx.x] = s2; __syncthreads();
  for (int off = 128; off > 0; off >>= 1) {
    if (threadIdx.x < off) red[threadIdx.x] += red[threadIdx.x + off];
    __syncthreads();
  }
  float var = red[0] * (1.f / C_) - mean * mean;
  float inv = rsqrtf(var + 1e-5f);
  for (int c = threadIdx.x; c < C_; c += 256)
    out[(size_t)row * C_ + c] = f2bf((xr[c] - mean) * inv * g[c] + bb[c]);
}

// ---------------------------------------------------------------------------
// Router: logits = h2 @ W_router, softmax, top-2 -> dense combine weights
// One wave per token.
// ---------------------------------------------------------------------------
__global__ __launch_bounds__(32)
void router_kernel(const __bf16* __restrict__ h2, const float* __restrict__ Wr,
                   float* __restrict__ w_moe, float* __restrict__ counts) {
  const int n = blockIdx.x, lane = threadIdx.x;
  float p[E_];
#pragma unroll
  for (int e = 0; e < E_; ++e) p[e] = 0.f;
  for (int c = lane; c < C_; c += 32) {
    float xv = bf2f(h2[(size_t)n * C_ + c]);
#pragma unroll
    for (int e = 0; e < E_; ++e) p[e] += xv * Wr[c * E_ + e];
  }
#pragma unroll
  for (int m = 16; m >= 1; m >>= 1)
#pragma unroll
    for (int e = 0; e < E_; ++e) p[e] += __shfl_xor(p[e], m, 32);
  if (lane == 0) {
    float mx = p[0];
    for (int e = 1; e < E_; ++e) mx = fmaxf(mx, p[e]);
    float pr[E_]; float sum = 0.f;
    for (int e = 0; e < E_; ++e) { pr[e] = __expf(p[e] - mx); sum += pr[e]; }
    for (int e = 0; e < E_; ++e) pr[e] /= sum;
    int i0 = 0;
    for (int e = 1; e < E_; ++e) if (pr[e] > pr[i0]) i0 = e;
    int i1 = (i0 == 0) ? 1 : 0;
    for (int e = 0; e < E_; ++e) if (e != i0 && pr[e] > pr[i1]) i1 = e;
    float t = pr[i0] + pr[i1] + 1e-9f;
    for (int e = 0; e < E_; ++e) w_moe[(size_t)n * E_ + e] = 0.f;
    w_moe[(size_t)n * E_ + i0] = pr[i0] / t;
    w_moe[(size_t)n * E_ + i1] = pr[i1] / t;
    atomicAdd(&counts[i0], 1.f);
    atomicAdd(&counts[i1], 1.f);
  }
}

__global__ void util_kernel(const float* __restrict__ counts, float* __restrict__ util) {
  if (threadIdx.x == 0 && blockIdx.x == 0) {
    float s = 0.f;
    for (int e = 0; e < E_; ++e) s += counts[e];
    for (int e = 0; e < E_; ++e) util[e] = counts[e] / (s + 1e-9f);
  }
}

__global__ void zero_kernel(float* __restrict__ p, long long n) {
  long long i = (long long)blockIdx.x * blockDim.x + threadIdx.x;
  long long st = (long long)gridDim.x * blockDim.x;
  for (; i < n; i += st) p[i] = 0.f;
}

__global__ void cvt_kernel(const float* __restrict__ in, __bf16* __restrict__ out,
                           long long n) {
  long long i = (long long)blockIdx.x * blockDim.x + threadIdx.x;
  long long st = (long long)gridDim.x * blockDim.x;
  for (; i < n; i += st) out[i] = f2bf(in[i]);
}

__global__ void add_kernel(const float* __restrict__ a, const float* __restrict__ b,
                           float* __restrict__ out, long long n) {
  long long i = (long long)blockIdx.x * blockDim.x + threadIdx.x;
  long long st = (long long)gridDim.x * blockDim.x;
  for (; i < n; i += st) out[i] = a[i] + b[i];
}

// ---------------------------------------------------------------------------
extern "C" void kernel_launch(void* const* d_in, const int* in_sizes, int n_in,
                              void* d_out, int out_size, void* d_ws, size_t ws_size,
                              hipStream_t stream) {
  (void)in_sizes; (void)n_in; (void)out_size; (void)ws_size;
  const float* x        = (const float*)d_in[0];
  const float* ln1_g    = (const float*)d_in[1];
  const float* ln1_b    = (const float*)d_in[2];
  const float* W_qkv    = (const float*)d_in[3];
  const float* b_qkv    = (const float*)d_in[4];
  const float* W_proj   = (const float*)d_in[5];
  const float* b_proj   = (const float*)d_in[6];
  const float* ln2_g    = (const float*)d_in[7];
  const float* ln2_b    = (const float*)d_in[8];
  const float* W_router = (const float*)d_in[9];
  const float* W1       = (const float*)d_in[10];
  const float* b1       = (const float*)d_in[11];
  const float* W2       = (const float*)d_in[12];
  const float* b2       = (const float*)d_in[13];

  float* out_x    = (float*)d_out;
  float* out_util = out_x + (size_t)N_ * C_;

  char* ws = (char*)d_ws;
  size_t off = 0;
  auto alloc = [&](size_t bytes) -> void* {
    void* p = ws + off;
    off += (bytes + 255) & ~(size_t)255;
    return p;
  };
  __bf16* h_bf    = (__bf16*)alloc((size_t)N_ * C_ * 2);
  __bf16* Wq_bf   = (__bf16*)alloc((size_t)C_ * 3 * C_ * 2);
  __bf16* qkv_bf  = (__bf16*)alloc((size_t)N_ * 3 * C_ * 2);
  __bf16* y_bf    = (__bf16*)alloc((size_t)N_ * C_ * 2);
  __bf16* Wp_bf   = (__bf16*)alloc((size_t)C_ * C_ * 2);
  float*  x1      = (float*) alloc((size_t)N_ * C_ * 4);
  __bf16* h2_bf   = (__bf16*)alloc((size_t)N_ * C_ * 2);
  __bf16* W1_bf   = (__bf16*)alloc((size_t)C_ * HID_ * 2);   // per-expert reuse
  __bf16* W2_bf   = (__bf16*)alloc((size_t)HID_ * C_ * 2);   // per-expert reuse
  __bf16* hid_bf  = (__bf16*)alloc((size_t)N_ * HID_ * 2);
  float*  w_moe   = (float*) alloc((size_t)N_ * E_ * 4);
  float*  counts  = (float*) alloc(64);
  float*  moe_out = (float*) alloc((size_t)N_ * C_ * 4);

  // --- attention branch ---
  cvt_kernel<<<2048, 256, 0, stream>>>(W_qkv, Wq_bf, (long long)C_ * 3 * C_);
  cvt_kernel<<<2048, 256, 0, stream>>>(W_proj, Wp_bf, (long long)C_ * C_);
  ln_pack_kernel<<<N_, 256, 0, stream>>>(x, ln1_g, ln1_b, h_bf);
  gemm_bf16_kernel<0><<<dim3(3 * C_ / BN, N_ / BM), 128, 0, stream>>>(
      h_bf, C_, Wq_bf, 3 * C_, b_qkv, nullptr, nullptr, 0,
      qkv_bf, nullptr, 3 * C_, N_, 3 * C_, C_);
  attn_kernel<<<dim3(T_ / 64, H_, B_), 128, 0, stream>>>(qkv_bf, y_bf);
  gemm_bf16_kernel<1><<<dim3(C_ / BN, N_ / BM), 128, 0, stream>>>(
      y_bf, C_, Wp_bf, C_, b_proj, x, nullptr, 0,
      nullptr, x1, C_, N_, C_, C_);

  // --- MoE branch ---
  ln_pack_kernel<<<N_, 256, 0, stream>>>(x1, ln2_g, ln2_b, h2_bf);
  zero_kernel<<<1, 32, 0, stream>>>(counts, E_);
  router_kernel<<<N_, 32, 0, stream>>>(h2_bf, W_router, w_moe, counts);
  util_kernel<<<1, 32, 0, stream>>>(counts, out_util);
  zero_kernel<<<2048, 256, 0, stream>>>(moe_out, (long long)N_ * C_);

  for (int e = 0; e < E_; ++e) {
    cvt_kernel<<<2048, 256, 0, stream>>>(W1 + (size_t)e * C_ * HID_, W1_bf,
                                         (long long)C_ * HID_);
    cvt_kernel<<<2048, 256, 0, stream>>>(W2 + (size_t)e * HID_ * C_, W2_bf,
                                         (long long)HID_ * C_);
    gemm_bf16_kernel<2><<<dim3(HID_ / BN, N_ / BM), 128, 0, stream>>>(
        h2_bf, C_, W1_bf, HID_, b1 + (size_t)e * HID_, nullptr, nullptr, 0,
        hid_bf, nullptr, HID_, N_, HID_, C_);
    gemm_bf16_kernel<3><<<dim3(C_ / BN, N_ / BM), 128, 0, stream>>>(
        hid_bf, HID_, W2_bf, C_, b2 + (size_t)e * C_, nullptr, w_moe + e, E_,
        nullptr, moe_out, C_, N_, C_, HID_);
  }

  add_kernel<<<2048, 256, 0, stream>>>(x1, moe_out, out_x, (long long)N_ * C_);
}
